// SS2D_89910845375197
// MI455X (gfx1250) — matrix operand order repaired and checked by reading, not compiled, not run on hardware
//
#include <hip/hip_runtime.h>
#include <cstdint>
#include <cstddef>

typedef __attribute__((ext_vector_type(16))) _Float16 v16h;
typedef __attribute__((ext_vector_type(8)))  _Float16 v8h;
typedef __attribute__((ext_vector_type(8)))  float    v8f;

#define DMODEL 192
#define DINNER 384
#define DSTATE 16
#define DTRANK 12
#define NBATCH 4
#define NSEQ   1024
#define BN     4096   // NBATCH*NSEQ

// Direction index mapping (involutions, shared by gather and scatter):
//   dir0: identity     dir1: flip      dir2: row<->col (H=W=32)   dir3: flip+transpose
__device__ __forceinline__ int map_l(int l, int dir){
  switch(dir){
    case 1:  return 1023 - l;
    case 2:  return ((l & 31) << 5) | (l >> 5);
    case 3:  { int t = 1023 - l; return ((t & 31) << 5) | (t >> 5); }
    default: return l;
  }
}

// -------------------- f32 -> f16 weight conversion / fills --------------------
__global__ void cvt_f16_kernel(_Float16* __restrict__ dst, const float* __restrict__ src, int n){
  int i = blockIdx.x * 256 + threadIdx.x;
  if (i < n) dst[i] = (_Float16)src[i];
}
__global__ void zero_h_kernel(_Float16* __restrict__ dst, int n){
  int i = blockIdx.x * 256 + threadIdx.x;
  if (i < n) dst[i] = (_Float16)0.f;
}
// pad each row of kin values to kout (zeros beyond kin) -- used for Wdt (K=12 -> 64)
__global__ void cvt_pad_kernel(_Float16* __restrict__ dst, const float* __restrict__ src,
                               int rows, int kin, int kout){
  int i = blockIdx.x * 256 + threadIdx.x;
  if (i >= rows * kout) return;
  int r = i / kout, k = i - r * kout;
  dst[i] = (k < kin) ? (_Float16)src[r * kin + k] : (_Float16)0.f;
}
// pad row count per direction block: (nd, rin, K) -> (nd, rout, K) zero rows beyond rin
// used for m_xp_w (44 rows -> 48) so GEMM N is always a multiple of 16 (no guards).
__global__ void cvt_pad_rows_kernel(_Float16* __restrict__ dst, const float* __restrict__ src,
                                    int nd, int rin, int rout, int K){
  int i = blockIdx.x * 256 + threadIdx.x;
  if (i >= nd * rout * K) return;
  int d = i / (rout * K);
  int rem = i - d * rout * K;
  int r = rem / K, k = rem - r * K;
  dst[i] = (r < rin) ? (_Float16)src[((size_t)d * rin + r) * K + k] : (_Float16)0.f;
}

// -------------------- input layernorm (B,192,1024) -> (B,1024,192) --------------------
__global__ void ln_in_kernel(const float* __restrict__ x, const float* __restrict__ g,
                             const float* __restrict__ b, float* __restrict__ xs,
                             _Float16* __restrict__ xsh){
  int lane = threadIdx.x & 31, wid = threadIdx.x >> 5;
  int r = blockIdx.x * 8 + wid;           // r = batch*1024 + n
  int bb = r >> 10, n = r & 1023;
  float v[6], s = 0.f, s2 = 0.f;
  #pragma unroll
  for (int j = 0; j < 6; ++j){
    int c = j * 32 + lane;
    v[j] = x[((size_t)(bb * DMODEL + c)) * NSEQ + n];
    s += v[j]; s2 += v[j] * v[j];
  }
  #pragma unroll
  for (int m = 16; m > 0; m >>= 1){ s += __shfl_xor(s, m, 32); s2 += __shfl_xor(s2, m, 32); }
  float mean = s * (1.f / DMODEL);
  float var  = s2 * (1.f / DMODEL) - mean * mean;
  float rstd = rsqrtf(var + 1e-5f);
  #pragma unroll
  for (int j = 0; j < 6; ++j){
    int c = j * 32 + lane;
    float o = (v[j] - mean) * rstd * g[c] + b[c];
    xs [(size_t)r * DMODEL + c] = o;
    xsh[(size_t)r * DMODEL + c] = (_Float16)o;
  }
}

// -------------------- final layernorm + residual, transpose to (B,192,H,W) --------------------
__global__ void ln_out_kernel(const float* __restrict__ yin, const float* __restrict__ g,
                              const float* __restrict__ b, const float* __restrict__ resid,
                              float* __restrict__ out){
  int lane = threadIdx.x & 31, wid = threadIdx.x >> 5;
  int r = blockIdx.x * 8 + wid;
  int bb = r >> 10, n = r & 1023;
  float v[6], s = 0.f, s2 = 0.f;
  #pragma unroll
  for (int j = 0; j < 6; ++j){
    int c = j * 32 + lane;
    v[j] = yin[(size_t)r * DMODEL + c];
    s += v[j]; s2 += v[j] * v[j];
  }
  #pragma unroll
  for (int m = 16; m > 0; m >>= 1){ s += __shfl_xor(s, m, 32); s2 += __shfl_xor(s2, m, 32); }
  float mean = s * (1.f / DMODEL);
  float var  = s2 * (1.f / DMODEL) - mean * mean;
  float rstd = rsqrtf(var + 1e-5f);
  #pragma unroll
  for (int j = 0; j < 6; ++j){
    int c = j * 32 + lane;
    float o = (v[j] - mean) * rstd * g[c] + b[c] + resid[(size_t)r * DMODEL + c];
    out[((size_t)(bb * DMODEL + c)) * NSEQ + n] = o;
  }
}

// -------------------- WMMA GEMM pipelined core (branch-free steady state) --------------------
__device__ __forceinline__ v16h make_av(const v8h& lo, const v8h& hi){
  v16h av;
  #pragma unroll
  for (int i = 0; i < 8; ++i){ av[i] = lo[i]; av[i + 8] = hi[i]; }
  return av;
}

template<bool HAS2>
__device__ __forceinline__ void gemm_core(const _Float16* __restrict__ arow,
                                          const _Float16* __restrict__ brow0,
                                          const _Float16* __restrict__ brow1,
                                          int K, v8f& acc0, v8f& acc1){
  // K % 64 == 0 guaranteed by padded operands.
  v8h  aloP = *(const v8h*)(arow);
  v8h  ahiP = *(const v8h*)(arow + 8);
  v16h bv0P = *(const v16h*)(brow0);
  v16h bv1P = bv0P;
  if (HAS2) bv1P = *(const v16h*)(brow1);

  int kk = 0;
  for (; kk + 64 < K; kk += 64){
    // odd chunk (Q) loads -- always valid in steady state
    v8h  aloQ = *(const v8h*)(arow + kk + 32);
    v8h  ahiQ = *(const v8h*)(arow + kk + 40);
    v16h bv0Q = *(const v16h*)(brow0 + kk + 32);
    v16h bv1Q = bv0Q;
    if (HAS2) bv1Q = *(const v16h*)(brow1 + kk + 32);
    __builtin_prefetch(arow + kk + 96, 0, 3);                    // global_prefetch_b8
    {
      v16h av = make_av(aloP, ahiP);
      acc0 = __builtin_amdgcn_wmma_f32_16x16x32_f16(false, av, false, bv0P, (short)0, acc0,
                                                    false, false);
      if (HAS2)
        acc1 = __builtin_amdgcn_wmma_f32_16x16x32_f16(false, av, false, bv1P, (short)0, acc1,
                                                      false, false);
    }
    // next even chunk (P) loads -- always valid in steady state
    aloP = *(const v8h*)(arow + kk + 64);
    ahiP = *(const v8h*)(arow + kk + 72);
    bv0P = *(const v16h*)(brow0 + kk + 64);
    if (HAS2) bv1P = *(const v16h*)(brow1 + kk + 64);
    {
      v16h av = make_av(aloQ, ahiQ);
      acc0 = __builtin_amdgcn_wmma_f32_16x16x32_f16(false, av, false, bv0Q, (short)0, acc0,
                                                    false, false);
      if (HAS2)
        acc1 = __builtin_amdgcn_wmma_f32_16x16x32_f16(false, av, false, bv1Q, (short)0, acc1,
                                                      false, false);
    }
  }
  // tail iteration (kk == K - 64): no next-P loads
  {
    v8h  aloQ = *(const v8h*)(arow + kk + 32);
    v8h  ahiQ = *(const v8h*)(arow + kk + 40);
    v16h bv0Q = *(const v16h*)(brow0 + kk + 32);
    v16h bv1Q = bv0Q;
    if (HAS2) bv1Q = *(const v16h*)(brow1 + kk + 32);
    {
      v16h av = make_av(aloP, ahiP);
      acc0 = __builtin_amdgcn_wmma_f32_16x16x32_f16(false, av, false, bv0P, (short)0, acc0,
                                                    false, false);
      if (HAS2)
        acc1 = __builtin_amdgcn_wmma_f32_16x16x32_f16(false, av, false, bv1P, (short)0, acc1,
                                                      false, false);
    }
    {
      v16h av = make_av(aloQ, ahiQ);
      acc0 = __builtin_amdgcn_wmma_f32_16x16x32_f16(false, av, false, bv0Q, (short)0, acc0,
                                                    false, false);
      if (HAS2)
        acc1 = __builtin_amdgcn_wmma_f32_16x16x32_f16(false, av, false, bv1Q, (short)0, acc1,
                                                      false, false);
    }
  }
}

// -------------------- generic WMMA GEMM: C[M,N] = A[M,K] * W[N,K]^T --------------------
// One wave per 16x32 output tile-pair (shared A fragment, two accumulators).
// Requirements: K % 64 == 0, M % 16 == 0, N % 16 == 0 (operands pre-padded).
// act: 0=none 1=gelu(exact) 2=softplus
// Oh epilogue: store f16 at [maprow * ldh + hcol0 + col] when col<hcolmax, zero-fill to hzfill.
__global__ void gemm_f16_wmma(const _Float16* __restrict__ A, int lda,
                              const _Float16* __restrict__ W, int ldb,
                              const float* __restrict__ bias,
                              int K, int Nt, int Np, int act,
                              float* __restrict__ Of, int ldo,
                              _Float16* __restrict__ Oh, int ldh, int hcol0, int hcolmax, int hzfill,
                              int rowmap, int pairs){
  int wid  = __builtin_amdgcn_readfirstlane(threadIdx.x >> 5);   // provably wave-uniform
  int lane = threadIdx.x & 31;
  int t = blockIdx.x * 4 + wid;
  if (t >= pairs) return;                       // scalar branch
  int tm = t / Np, tp = t - tm * Np;
  int tn0 = tp * 2;
  bool has2 = (tn0 + 1) < Nt;                   // scalar, hoisted out of the K loop
  int row0 = tm << 4;
  int col0 = tn0 << 4;
  int m  = lane & 15;
  int gl = lane >> 4;
  int n  = lane & 15;
  const _Float16* arow  = A + (size_t)(row0 + m) * lda + 16 * gl;  // 16 contiguous K at 16g
  const _Float16* brow0 = W + (size_t)(col0 + n) * ldb + 16 * gl;  // 16 contiguous K at 16g
  const _Float16* brow1 = brow0 + (size_t)16 * ldb;

  v8f acc0 = {0.f, 0.f, 0.f, 0.f, 0.f, 0.f, 0.f, 0.f};
  v8f acc1 = {0.f, 0.f, 0.f, 0.f, 0.f, 0.f, 0.f, 0.f};
  if (has2) gemm_core<true >(arow, brow0, brow1, K, acc0, acc1);
  else      gemm_core<false>(arow, brow0, brow1, K, acc0, acc1);

  #pragma unroll
  for (int tt = 0; tt < 2; ++tt){
    if (tt && !has2) break;                      // scalar
    int col = col0 + tt * 16 + n;
    const v8f& acc = tt ? acc1 : acc0;
    float bval = (bias != nullptr) ? bias[col] : 0.f;
    #pragma unroll
    for (int i = 0; i < 8; ++i){
      int row = row0 + gl * 8 + i;
      float v = acc[i] + bval;
      if (act == 1)      v = 0.5f * v * (1.f + erff(v * 0.70710678118654752f));
      else if (act == 2) v = (v > 20.f) ? v : log1pf(__expf(v));
      if (Of) Of[(size_t)row * ldo + col] = v;
      if (Oh){
        int hrow = row;
        if (rowmap > 0){ int bb = row >> 10, l = row & 1023; hrow = (bb << 10) + map_l(l, rowmap); }
        if (col < hcolmax)     Oh[(size_t)hrow * ldh + hcol0 + col] = (_Float16)v;
        else if (col < hzfill) Oh[(size_t)hrow * ldh + hcol0 + col] = (_Float16)0.f;
      }
    }
  }
}

// -------------------- gather directional sequence (f16) --------------------
__global__ void seq_build_kernel(_Float16* __restrict__ seq, const float* __restrict__ xp, int dir){
  int idx = blockIdx.x * 256 + threadIdx.x;      // over BN*DMODEL
  int c = idx % DMODEL;
  int r = idx / DMODEL;
  int l = r & 1023, bb = r >> 10;
  int lp = map_l(l, dir);
  seq[idx] = (_Float16)xp[((size_t)(bb << 10) + lp) * DMODEL + c];
}

// -------------------- depthwise causal conv1d (k=4) + SiLU --------------------
__global__ void conv_silu_kernel(const float* __restrict__ xz, const float* __restrict__ cw,
                                 const float* __restrict__ cb, float* __restrict__ xsc,
                                 _Float16* __restrict__ xsch){
  int idx = blockIdx.x * 256 + threadIdx.x;      // over BN*DINNER
  int c  = idx % DINNER;
  int bl = idx / DINNER;
  int l  = bl & 1023;
  size_t rb = (size_t)(bl - l) * (2 * DINNER);   // (b*1024) * 768
  float acc = cb[c];
  #pragma unroll
  for (int j = 0; j < 4; ++j){
    int ll = l - 3 + j;
    if (ll >= 0) acc += cw[c * 4 + j] * xz[rb + (size_t)ll * (2 * DINNER) + c];
  }
  float s = acc / (1.f + __expf(-acc));
  xsc [idx] = s;
  xsch[idx] = (_Float16)s;
}

// -------------------- selective scan: h = exp(dt*A)h + dt*B*x ; y = <h,C> --------------------
// grid (batch, channel-group of 128); 16 states per channel live in VGPRs.
// B_t/C_t (shared across channels) staged through LDS in 4-step chunks.
__global__ void scan_kernel(const float* __restrict__ dtf, const float* __restrict__ dbl,
                            const float* __restrict__ xsc, const float* __restrict__ xz,
                            const float* __restrict__ Alog, const float* __restrict__ Dp,
                            _Float16* __restrict__ yh){
  const int bb  = blockIdx.x;
  const int tid = threadIdx.x;
  const int c   = blockIdx.y * 128 + tid;
  __shared__ float sB[4][DSTATE];
  __shared__ float sC[4][DSTATE];
  float Ac[DSTATE];
  #pragma unroll
  for (int e = 0; e < DSTATE; ++e) Ac[e] = -__expf(Alog[c * DSTATE + e]);
  const float Dv = Dp[c];
  float h[DSTATE];
  #pragma unroll
  for (int e = 0; e < DSTATE; ++e) h[e] = 0.f;
  const size_t rowbase = (size_t)bb * NSEQ;

  for (int l0 = 0; l0 < NSEQ; l0 += 4){
    __syncthreads();
    { int step = tid >> 5, j = tid & 31;
      float v = dbl[(rowbase + l0 + step) * 48 + DTRANK + j];
      if (j < DSTATE) sB[step][j] = v; else sC[step][j - DSTATE] = v; }
    __syncthreads();
    #pragma unroll
    for (int s = 0; s < 4; ++s){
      size_t idx = (rowbase + l0 + s) * DINNER + c;
      float dtv = dtf[idx];
      float xv  = xsc[idx];
      float acc = 0.f;
      #pragma unroll
      for (int e = 0; e < DSTATE; ++e){
        float dA = __expf(dtv * Ac[e]);
        h[e] = dA * h[e] + dtv * sB[s][e] * xv;
        acc += h[e] * sC[s][e];
      }
      float zv = xz[(rowbase + l0 + s) * (2 * DINNER) + DINNER + c];
      float yv = (acc + xv * Dv) * (zv / (1.f + __expf(-zv)));
      yh[idx] = (_Float16)yv;
    }
  }
}

// -------------------- host launcher --------------------
static void launch_gemm(hipStream_t st,
                        const _Float16* A, int lda, const _Float16* W, int ldb,
                        const float* bias, int M, int N, int K, int act,
                        float* Of, int ldo,
                        _Float16* Oh, int ldh, int hcol0, int hcolmax, int hzfill,
                        int rowmap){
  int Mt = M / 16, Nt = N / 16;        // N must be a multiple of 16 (padded weights)
  int Np = (Nt + 1) / 2;               // tile pairs per row strip
  int pairs = Mt * Np;
  int blocks = (pairs + 3) / 4;
  gemm_f16_wmma<<<blocks, 128, 0, st>>>(A, lda, W, ldb, bias, K, Nt, Np, act,
                                        Of, ldo, Oh, ldh, hcol0, hcolmax, hzfill, rowmap, pairs);
}

extern "C" void kernel_launch(void* const* d_in, const int* in_sizes, int n_in,
                              void* d_out, int out_size, void* d_ws, size_t ws_size,
                              hipStream_t stream){
  (void)in_sizes; (void)n_in; (void)out_size; (void)ws_size;
  const float* x       = (const float*)d_in[0];
  const float* norm_g  = (const float*)d_in[1];
  const float* norm_b  = (const float*)d_in[2];
  const float* in_w    = (const float*)d_in[3];
  const float* in_b    = (const float*)d_in[4];
  const float* m_in_w  = (const float*)d_in[5];
  const float* m_conv_w= (const float*)d_in[6];
  const float* m_conv_b= (const float*)d_in[7];
  const float* m_xp_w  = (const float*)d_in[8];
  const float* m_dt_w  = (const float*)d_in[9];
  const float* m_dt_b  = (const float*)d_in[10];
  const float* m_Alog  = (const float*)d_in[11];
  const float* m_D     = (const float*)d_in[12];
  const float* m_out_w = (const float*)d_in[13];
  const float* f_w1    = (const float*)d_in[14];
  const float* f_b1    = (const float*)d_in[15];
  const float* f_w2    = (const float*)d_in[16];
  const float* f_b2    = (const float*)d_in[17];
  const float* o_w     = (const float*)d_in[18];
  const float* o_b     = (const float*)d_in[19];
  const float* on_g    = (const float*)d_in[20];
  const float* on_b    = (const float*)d_in[21];

  char* base = (char*)d_ws;
  auto alloc = [&](size_t bytes) -> char* {
    char* p = base; base += (bytes + 255) & ~(size_t)255; return p;
  };
  float*    xs_ln    = (float*)   alloc((size_t)BN * DMODEL * 4);
  _Float16* xs_ln_h  = (_Float16*)alloc((size_t)BN * DMODEL * 2);
  float*    xp       = (float*)   alloc((size_t)BN * DMODEL * 4);
  _Float16* seq_h    = (_Float16*)alloc((size_t)BN * DMODEL * 2);
  float*    xz       = (float*)   alloc((size_t)BN * 2 * DINNER * 4);
  float*    xsc      = (float*)   alloc((size_t)BN * DINNER * 4);
  _Float16* xsc_h    = (_Float16*)alloc((size_t)BN * DINNER * 2);
  float*    dbl      = (float*)   alloc((size_t)BN * 48 * 4);       // 44 cols (+4 pad), stride 48
  _Float16* dtraw_h  = (_Float16*)alloc((size_t)BN * 64 * 2);       // 12 cols zero-padded to 64
  float*    dtf      = (float*)   alloc((size_t)BN * DINNER * 4);
  _Float16* y_h      = (_Float16*)alloc((size_t)BN * DINNER * 2);
  _Float16* fused_h  = (_Float16*)alloc((size_t)BN * 4 * DMODEL * 2);
  _Float16* hdn_h    = (_Float16*)alloc((size_t)BN * 2 * DMODEL * 2);
  _Float16* ffn2_h   = (_Float16*)alloc((size_t)BN * DMODEL * 2);
  float*    outp     = (float*)   alloc((size_t)BN * DMODEL * 4);
  _Float16* in_w_h   = (_Float16*)alloc((size_t)DMODEL * DMODEL * 2);
  _Float16* m_in_w_h = (_Float16*)alloc((size_t)4 * 2 * DINNER * DMODEL * 2);
  _Float16* m_xp_w_h = (_Float16*)alloc((size_t)4 * 48 * DINNER * 2);   // rows padded 44 -> 48
  _Float16* m_dt_w_h = (_Float16*)alloc((size_t)4 * DINNER * 64 * 2);   // K padded 12 -> 64
  _Float16* m_out_w_h= (_Float16*)alloc((size_t)4 * DMODEL * DINNER * 2);
  _Float16* f_w1_h   = (_Float16*)alloc((size_t)2 * DMODEL * 4 * DMODEL * 2);
  _Float16* f_w2_h   = (_Float16*)alloc((size_t)DMODEL * 2 * DMODEL * 2);
  _Float16* o_w_h    = (_Float16*)alloc((size_t)DMODEL * DMODEL * 2);

  auto cvt = [&](_Float16* d, const float* s, int n){
    cvt_f16_kernel<<<(n + 255) / 256, 256, 0, stream>>>(d, s, n);
  };
  cvt(in_w_h,    in_w,    DMODEL * DMODEL);
  cvt(m_in_w_h,  m_in_w,  4 * 2 * DINNER * DMODEL);
  cvt(m_out_w_h, m_out_w, 4 * DMODEL * DINNER);
  cvt(f_w1_h,    f_w1,    2 * DMODEL * 4 * DMODEL);
  cvt(f_w2_h,    f_w2,    DMODEL * 2 * DMODEL);
  cvt(o_w_h,     o_w,     DMODEL * DMODEL);
  cvt_pad_rows_kernel<<<(4 * 48 * DINNER + 255) / 256, 256, 0, stream>>>(
      m_xp_w_h, m_xp_w, 4, 44, 48, DINNER);
  cvt_pad_kernel<<<(4 * DINNER * 64 + 255) / 256, 256, 0, stream>>>(m_dt_w_h, m_dt_w,
                                                                    4 * DINNER, DTRANK, 64);
  // dtraw columns [48,64) are never written by the xproj epilogue (tiles cover 48 cols):
  // zero the whole buffer once so the dt GEMM's padded K region reads zeros.
  zero_h_kernel<<<(BN * 64 + 255) / 256, 256, 0, stream>>>(dtraw_h, BN * 64);

  // 1) layernorm (+ transpose to row-major tokens)
  ln_in_kernel<<<BN / 8, 256, 0, stream>>>(x, norm_g, norm_b, xs_ln, xs_ln_h);

  // 2) in-proj GEMM: xp = xs_ln @ in_w^T + in_b
  launch_gemm(stream, xs_ln_h, DMODEL, in_w_h, DMODEL, in_b, BN, DMODEL, DMODEL, 0,
              xp, DMODEL, nullptr, 0, 0, 0, 0, 0);

  // 3) four directional mamba branches (buffers reused; scatter into fused_h)
  for (int dir = 0; dir < 4; ++dir){
    seq_build_kernel<<<(BN * DMODEL) / 256, 256, 0, stream>>>(seq_h, xp, dir);
    // xz = seq @ Win^T   (N=768)
    launch_gemm(stream, seq_h, DMODEL, m_in_w_h + (size_t)dir * 2 * DINNER * DMODEL, DMODEL,
                nullptr, BN, 2 * DINNER, DMODEL, 0, xz, 2 * DINNER,
                nullptr, 0, 0, 0, 0, 0);
    conv_silu_kernel<<<(BN * DINNER) / 256, 256, 0, stream>>>(
        xz, m_conv_w + (size_t)dir * DINNER * 4, m_conv_b + (size_t)dir * DINNER, xsc, xsc_h);
    // dbl = xs @ Wxp^T  (N padded 44->48; f32 stride 48; dt_raw cols [0,12) to f16, zeros to 48)
    launch_gemm(stream, xsc_h, DINNER, m_xp_w_h + (size_t)dir * 48 * DINNER, DINNER,
                nullptr, BN, 48, DINNER, 0, dbl, 48,
                dtraw_h, 64, 0, DTRANK, 48, 0);
    // dt = softplus(dt_raw @ Wdt^T + bdt)  (K zero-padded to 64)
    launch_gemm(stream, dtraw_h, 64, m_dt_w_h + (size_t)dir * DINNER * 64, 64,
                m_dt_b + (size_t)dir * DINNER, BN, DINNER, 64, 2, dtf, DINNER,
                nullptr, 0, 0, 0, 0, 0);
    scan_kernel<<<dim3(NBATCH, DINNER / 128), 128, 0, stream>>>(
        dtf, dbl, xsc, xz, m_Alog + (size_t)dir * DINNER * DSTATE,
        m_D + (size_t)dir * DINNER, y_h);
    // out-proj, scattered (row-permuted) into fused[:, dir*192 : (dir+1)*192] as f16
    launch_gemm(stream, y_h, DINNER, m_out_w_h + (size_t)dir * DMODEL * DINNER, DINNER,
                nullptr, BN, DMODEL, DINNER, 0, nullptr, 0,
                fused_h, 4 * DMODEL, dir * DMODEL, DMODEL, DMODEL, dir);
  }

  // 4) FFN: gelu(fused @ f_w1^T + b1) @ f_w2^T + b2
  launch_gemm(stream, fused_h, 4 * DMODEL, f_w1_h, 4 * DMODEL, f_b1, BN, 2 * DMODEL, 4 * DMODEL, 1,
              nullptr, 0, hdn_h, 2 * DMODEL, 0, 2 * DMODEL, 2 * DMODEL, 0);
  launch_gemm(stream, hdn_h, 2 * DMODEL, f_w2_h, 2 * DMODEL, f_b2, BN, DMODEL, 2 * DMODEL, 0,
              nullptr, 0, ffn2_h, DMODEL, 0, DMODEL, DMODEL, 0);

  // 5) output projection
  launch_gemm(stream, ffn2_h, DMODEL, o_w_h, DMODEL, o_b, BN, DMODEL, DMODEL, 0,
              outp, DMODEL, nullptr, 0, 0, 0, 0, 0);

  // 6) final LN + residual, transpose back to (B,192,H,W)
  ln_out_kernel<<<BN / 8, 256, 0, stream>>>(outp, on_g, on_b, xs_ln, (float*)d_out);
}
